// SelfAttention_22874995818737
// MI455X (gfx1250) — matrix operand order repaired and checked by reading, not compiled, not run on hardware
//
#include <hip/hip_runtime.h>
#include <hip/hip_bf16.h>

// Problem constants (match reference)
constexpr int Bc = 4;
constexpr int Sc = 2048;
constexpr int Dc = 1024;
constexpr int Hc = 16;
constexpr int HDc = 64;
constexpr int Mrows = Bc * Sc;           // 8192

typedef __bf16 v16bf __attribute__((ext_vector_type(16)));
typedef float  v8f   __attribute__((ext_vector_type(8)));

union Frag {
    v16bf v;
    uint4 q[2];
};

__device__ inline unsigned short f2bf(float f) {
    union { float f; unsigned u; } x;
    x.f = f;
    unsigned u = x.u + 0x7FFFu + ((x.u >> 16) & 1u);   // round-to-nearest-even
    return (unsigned short)(u >> 16);
}

__device__ inline v8f wmma_bf16(const Frag& a, const Frag& b, v8f c) {
    return __builtin_amdgcn_wmma_f32_16x16x32_bf16(
        false, a.v, false, b.v, (short)0, c, false, false);
}

// ---------------------------------------------------------------------------
// Elementwise f32 -> bf16
// ---------------------------------------------------------------------------
__global__ void cvt_f32_bf16(const float* __restrict__ in,
                             unsigned short* __restrict__ out, int n) {
    int i = blockIdx.x * blockDim.x + threadIdx.x;
    int stride = gridDim.x * blockDim.x;
    for (; i < n; i += stride) out[i] = f2bf(in[i]);
}

// ---------------------------------------------------------------------------
// GEMM:  C[M,N] = A[M,K] @ W[N,K]^T + bias    (torch Linear semantics)
// A, W bf16 row-major.  One wave computes a 32x32 tile (2x2 WMMA tiles) so
// each k-step does 4 WMMAs on 4 fragment loads (2 A + 2 B).
// Block = 256 thr = 8 waves arranged 2(M) x 4(N)  -> 64 x 128 per block.
// MODE 0: bf16 out row-major [M,N]
// MODE 1: f32  out row-major [M,N]
// MODE 2: bf16 out head-transposed: out[(b*N + n)*S + s]   (V^T layout)
// ---------------------------------------------------------------------------
template <int MODE>
__global__ __launch_bounds__(256)
void gemm_wmma(const unsigned short* __restrict__ A,
               const unsigned short* __restrict__ W,
               const float* __restrict__ bias,
               void* __restrict__ outp,
               int M, int N, int K) {
    const int lane = threadIdx.x & 31;
    const int wave = threadIdx.x >> 5;
    const int wn = wave & 3, wm = wave >> 2;
    const int m0 = blockIdx.y * 64 + wm * 32;
    const int n0 = blockIdx.x * 128 + wn * 32;
    const int lh  = lane >> 4;       // lane half
    const int lid = lane & 15;

    const unsigned short* arow0 = A + (size_t)(m0 + lid) * K + 8 * lh;
    const unsigned short* arow1 = arow0 + (size_t)16 * K;
    const unsigned short* brow0 = W + (size_t)(n0 + lid) * K + 8 * lh;
    const unsigned short* brow1 = brow0 + (size_t)16 * K;

    v8f acc00 = {0.f,0.f,0.f,0.f,0.f,0.f,0.f,0.f};
    v8f acc01 = acc00, acc10 = acc00, acc11 = acc00;

#pragma unroll 2
    for (int kb = 0; kb < K; kb += 32) {
        Frag fa0, fa1, fb0, fb1;
        fa0.q[0] = *(const uint4*)(arow0 + kb);
        fa0.q[1] = *(const uint4*)(arow0 + kb + 16);
        fa1.q[0] = *(const uint4*)(arow1 + kb);
        fa1.q[1] = *(const uint4*)(arow1 + kb + 16);
        fb0.q[0] = *(const uint4*)(brow0 + kb);
        fb0.q[1] = *(const uint4*)(brow0 + kb + 16);
        fb1.q[0] = *(const uint4*)(brow1 + kb);
        fb1.q[1] = *(const uint4*)(brow1 + kb + 16);
        acc00 = wmma_bf16(fa0, fb0, acc00);
        acc01 = wmma_bf16(fa0, fb1, acc01);
        acc10 = wmma_bf16(fa1, fb0, acc10);
        acc11 = wmma_bf16(fa1, fb1, acc11);
    }

    const int na = n0 + lid;
    const int nb = na + 16;
    const float bva = bias[na];
    const float bvb = bias[nb];

#pragma unroll
    for (int r = 0; r < 8; ++r) {
        int ma = m0 + 8 * lh + r;
        int mb = ma + 16;
        float v00 = acc00[r] + bva;   // (ma, na)
        float v01 = acc01[r] + bvb;   // (ma, nb)
        float v10 = acc10[r] + bva;   // (mb, na)
        float v11 = acc11[r] + bvb;   // (mb, nb)
        if (MODE == 1) {
            float* out = (float*)outp;
            out[(size_t)ma * N + na] = v00;
            out[(size_t)ma * N + nb] = v01;
            out[(size_t)mb * N + na] = v10;
            out[(size_t)mb * N + nb] = v11;
        } else if (MODE == 0) {
            unsigned short* out = (unsigned short*)outp;
            out[(size_t)ma * N + na] = f2bf(v00);
            out[(size_t)ma * N + nb] = f2bf(v01);
            out[(size_t)mb * N + na] = f2bf(v10);
            out[(size_t)mb * N + nb] = f2bf(v11);
        } else {
            unsigned short* out = (unsigned short*)outp;
            int ba = ma >> 11, sa = ma & (Sc - 1);
            int bb = mb >> 11, sb = mb & (Sc - 1);
            out[((size_t)ba * N + na) * Sc + sa] = f2bf(v00);
            out[((size_t)ba * N + nb) * Sc + sa] = f2bf(v01);
            out[((size_t)bb * N + na) * Sc + sb] = f2bf(v10);
            out[((size_t)bb * N + nb) * Sc + sb] = f2bf(v11);
        }
    }
}

// ---------------------------------------------------------------------------
// Flash attention: one wave handles a 16-query tile for one (b,h).
// Processes 64 keys per iteration (4 score tiles).  Row max reduced across
// lanes each chunk; row SUM kept as per-lane partials (corr is lane-uniform)
// and reduced once after the loop.
// Q,K: bf16 [B*S, D] row-major.  Vt: bf16 [B,H,HD,S].  Out: bf16 [B*S, D].
// ---------------------------------------------------------------------------
__global__ __launch_bounds__(128)
void attn_wmma(const unsigned short* __restrict__ Q,
               const unsigned short* __restrict__ Kb,
               const unsigned short* __restrict__ Vt,
               unsigned short* __restrict__ Ob) {
    __shared__ __align__(16) unsigned short plds[4][16 * 64];

    const int lane = threadIdx.x & 31;
    const int wave = threadIdx.x >> 5;
    const int w = blockIdx.x * 4 + wave;
    const int bh = w >> 7;              // / (S/16)
    const int qt = w & 127;
    const int b = bh >> 4, h = bh & 15;
    const int q0 = qt * 16;
    const int lh  = lane >> 4;
    const int lid = lane & 15;

    // Q tile as two A-fragments (hd 0..31, 32..63)
    const unsigned short* qrow =
        Q + ((size_t)(b * Sc + q0 + lid) * Dc + h * HDc) + 8 * lh;
    Frag a0, a1;
    a0.q[0] = *(const uint4*)(qrow);
    a0.q[1] = *(const uint4*)(qrow + 16);
    a1.q[0] = *(const uint4*)(qrow + 32);
    a1.q[1] = *(const uint4*)(qrow + 48);

    const unsigned short* kbase = Kb + (size_t)(b * Sc) * Dc + h * HDc;
    const unsigned short* vbase = Vt + (size_t)(b * Hc + h) * HDc * Sc;

    v8f o0 = {0.f,0.f,0.f,0.f,0.f,0.f,0.f,0.f};
    v8f o1 = o0, o2 = o0, o3 = o0;
    float mrow[8], lsum[8];
#pragma unroll
    for (int r = 0; r < 8; ++r) { mrow[r] = -1e30f; lsum[r] = 0.f; }

    const float cexp = 0.125f * 1.44269504088896340736f;  // scale * log2(e)

    for (int kb = 0; kb < Sc; kb += 64) {
        // ----- scores: four 16x16 tiles over keys [kb..kb+63]
        v8f s[4];
#pragma unroll
        for (int t = 0; t < 4; ++t) {
            const unsigned short* kr =
                kbase + (size_t)(kb + 16 * t + lid) * Dc + 8 * lh;
            Frag k0, k1;
            k0.q[0] = *(const uint4*)(kr);      k0.q[1] = *(const uint4*)(kr + 16);
            k1.q[0] = *(const uint4*)(kr + 32); k1.q[1] = *(const uint4*)(kr + 48);
            v8f st = {0.f,0.f,0.f,0.f,0.f,0.f,0.f,0.f};
            st = wmma_bf16(a0, k0, st);
            st = wmma_bf16(a1, k1, st);
            s[t] = st;
        }

        // ----- online softmax update (row = 8*lh + r, col = lid within tile)
#pragma unroll
        for (int r = 0; r < 8; ++r) {
            float t0 = s[0][r], t1 = s[1][r], t2 = s[2][r], t3 = s[3][r];
            float mx = fmaxf(fmaxf(t0, t1), fmaxf(t2, t3));
#pragma unroll
            for (int msk = 1; msk <= 8; msk <<= 1)
                mx = fmaxf(mx, __shfl_xor(mx, msk, 32));
            float nm = fmaxf(mrow[r], mx);
            float corr = exp2f((mrow[r] - nm) * cexp);
            mrow[r] = nm;
            float p0 = exp2f((t0 - nm) * cexp);
            float p1 = exp2f((t1 - nm) * cexp);
            float p2 = exp2f((t2 - nm) * cexp);
            float p3 = exp2f((t3 - nm) * cexp);
            // per-lane partial sum; cross-lane reduction deferred to the end
            lsum[r] = lsum[r] * corr + ((p0 + p1) + (p2 + p3));
            o0[r] *= corr; o1[r] *= corr; o2[r] *= corr; o3[r] *= corr;
            int m = 8 * lh + r;
            plds[wave][m * 64 + lid]      = f2bf(p0);
            plds[wave][m * 64 + 16 + lid] = f2bf(p1);
            plds[wave][m * 64 + 32 + lid] = f2bf(p2);
            plds[wave][m * 64 + 48 + lid] = f2bf(p3);
        }

        // P is per-wave private in LDS; same-wave store->load ordering
        asm volatile("s_wait_dscnt 0" ::: "memory");

        // ----- P as two A-fragments (keys 0..31 and 32..63 of the chunk)
        Frag pa0, pa1;
        pa0.q[0] = *(const uint4*)&plds[wave][lid * 64 + 8 * lh];
        pa0.q[1] = *(const uint4*)&plds[wave][lid * 64 + 8 * lh + 16];
        pa1.q[0] = *(const uint4*)&plds[wave][lid * 64 + 32 + 8 * lh];
        pa1.q[1] = *(const uint4*)&plds[wave][lid * 64 + 32 + 8 * lh + 16];

        // ----- PV: 4 output tiles x 2 key sub-chunks
#pragma unroll
        for (int t = 0; t < 4; ++t) {
            const unsigned short* vr =
                vbase + (size_t)(16 * t + lid) * Sc + kb + 8 * lh;
            Frag v0, v1;
            v0.q[0] = *(const uint4*)(vr);      v0.q[1] = *(const uint4*)(vr + 16);
            v1.q[0] = *(const uint4*)(vr + 32); v1.q[1] = *(const uint4*)(vr + 48);
            if (t == 0) { o0 = wmma_bf16(pa0, v0, o0); o0 = wmma_bf16(pa1, v1, o0); }
            if (t == 1) { o1 = wmma_bf16(pa0, v0, o1); o1 = wmma_bf16(pa1, v1, o1); }
            if (t == 2) { o2 = wmma_bf16(pa0, v0, o2); o2 = wmma_bf16(pa1, v1, o2); }
            if (t == 3) { o3 = wmma_bf16(pa0, v0, o3); o3 = wmma_bf16(pa1, v1, o3); }
        }
    }

    // ----- final cross-lane sum reduction, normalize, store bf16 [B*S, D]
#pragma unroll
    for (int r = 0; r < 8; ++r) {
        float sum = lsum[r];
#pragma unroll
        for (int msk = 1; msk <= 8; msk <<= 1)
            sum += __shfl_xor(sum, msk, 32);
        float inv = 1.0f / sum;
        int m = q0 + 8 * lh + r;
        size_t base = (size_t)(b * Sc + m) * Dc + h * HDc + lid;
        Ob[base +  0] = f2bf(o0[r] * inv);
        Ob[base + 16] = f2bf(o1[r] * inv);
        Ob[base + 32] = f2bf(o2[r] * inv);
        Ob[base + 48] = f2bf(o3[r] * inv);
    }
}

// ---------------------------------------------------------------------------
// Host launch
// ---------------------------------------------------------------------------
extern "C" void kernel_launch(void* const* d_in, const int* in_sizes, int n_in,
                              void* d_out, int out_size, void* d_ws, size_t ws_size,
                              hipStream_t stream) {
    const float* x  = (const float*)d_in[0];
    const float* Wq = (const float*)d_in[1];
    const float* bq = (const float*)d_in[2];
    const float* Wk = (const float*)d_in[3];
    const float* bk = (const float*)d_in[4];
    const float* Wv = (const float*)d_in[5];
    const float* bv = (const float*)d_in[6];
    const float* Wo = (const float*)d_in[7];
    const float* bo = (const float*)d_in[8];
    float* out = (float*)d_out;

    // Workspace layout (bf16 = unsigned short); total 72 MiB
    unsigned short* ws = (unsigned short*)d_ws;
    const size_t MD = (size_t)Mrows * Dc;   // 8M elems
    const size_t DD = (size_t)Dc * Dc;      // 1M elems
    unsigned short* xb  = ws;               // [M,D]
    unsigned short* wqb = xb  + MD;
    unsigned short* wkb = wqb + DD;
    unsigned short* wvb = wkb + DD;
    unsigned short* wob = wvb + DD;
    unsigned short* Qb  = wob + DD;         // [M,D]
    unsigned short* Kbf = Qb  + MD;         // [M,D]
    unsigned short* Vtb = Kbf + MD;         // [B,H,HD,S]
    unsigned short* att = xb;               // alias: x dead after QKV GEMMs

    // 1) f32 -> bf16 conversions
    cvt_f32_bf16<<<2048, 256, 0, stream>>>(x,  xb,  (int)MD);
    cvt_f32_bf16<<<1024, 256, 0, stream>>>(Wq, wqb, (int)DD);
    cvt_f32_bf16<<<1024, 256, 0, stream>>>(Wk, wkb, (int)DD);
    cvt_f32_bf16<<<1024, 256, 0, stream>>>(Wv, wvb, (int)DD);
    cvt_f32_bf16<<<1024, 256, 0, stream>>>(Wo, wob, (int)DD);

    // 2) Q, K, V projections (WMMA)
    dim3 ggrid(Dc / 128, Mrows / 64);   // (8, 128)
    gemm_wmma<0><<<ggrid, 256, 0, stream>>>(xb, wqb, bq, Qb,  Mrows, Dc, Dc);
    gemm_wmma<0><<<ggrid, 256, 0, stream>>>(xb, wkb, bk, Kbf, Mrows, Dc, Dc);
    gemm_wmma<2><<<ggrid, 256, 0, stream>>>(xb, wvb, bv, Vtb, Mrows, Dc, Dc);

    // 3) Flash attention (WMMA, online softmax, 64 keys/iter)
    attn_wmma<<<(Bc * Hc * (Sc / 16)) / 4, 128, 0, stream>>>(Qb, Kbf, Vtb, att);

    // 4) Output projection, f32 result
    gemm_wmma<1><<<ggrid, 256, 0, stream>>>(att, wob, bo, out, Mrows, Dc, Dc);
}
